// EncoderRNN_309237645857
// MI455X (gfx1250) — matrix operand order repaired and checked by reading, not compiled, not run on hardware
//
#include <hip/hip_runtime.h>

// Tree-GRU encoder (bottom-up + top-down GRU scans) for MI455X / gfx1250.
// Strategy: latency-bound sequential scan -> one fused WMMA kernel per step,
// weights pre-packed once into bf16 (stays hot in 192MB L2), child-sum einsum
// replaced by O(L*B*H) parent scatter using td_parent_idx.
// Round 2: fast transcendentals (v_rcp_f32 / v_tanh_f32) on the per-step
// critical path instead of IEEE divide expansion + libm tanh.

#define Lc 256
#define Bc 128
#define Dc 512
#define Hc 512
#define KKc 1024   // D + H
#define NPc 2048   // packed gate columns: [rz combined | xn | hn]

typedef __attribute__((ext_vector_type(16))) __bf16     v16bf;
typedef __attribute__((ext_vector_type(8)))  float      v8f;
typedef __attribute__((ext_vector_type(4)))  unsigned   v4u;
typedef __attribute__((ext_vector_type(2)))  unsigned   v2u;

struct Frag32 { v4u lo, hi; };   // 32 bytes == one v16bf operand

__device__ __forceinline__ unsigned short f2bf(float f) {
  unsigned u = __float_as_uint(f);
  u += 0x7FFFu + ((u >> 16) & 1u);       // round-to-nearest-even
  return (unsigned short)(u >> 16);
}

// fast sigmoid: one v_exp_f32 + one v_rcp_f32 (trans ops, co-execute w/ VALU)
__device__ __forceinline__ float fast_sigmoid(float x) {
  return __builtin_amdgcn_rcpf(1.f + __expf(-x));
}

__device__ __forceinline__ float fast_tanh(float x) {
#if __has_builtin(__builtin_amdgcn_tanhf)
  return __builtin_amdgcn_tanhf(x);      // single v_tanh_f32
#else
  // tanh(x) = 1 - 2/(exp(2x)+1), exp/rcp trans ops only
  float e = __expf(2.f * x);
  return 1.f - 2.f * __builtin_amdgcn_rcpf(e + 1.f);
#endif
}

// ---------------------------------------------------------------- zero fill
__global__ void zero_f32(float* __restrict__ p, int n4) {
  int i = blockIdx.x * blockDim.x + threadIdx.x;
  if (i < n4) ((float4*)p)[i] = make_float4(0.f, 0.f, 0.f, 0.f);
}

// ------------------------------------------------- pack weights, bf16, N-major
// WpackT[n][k], n in [0,2048): cols 0..1023 = r,z of (Wx;Uh),
// 1024..1535 = Wx n-gate (h rows zero), 1536..2047 = Uh n-gate (x rows zero).
__global__ void pack_weights(const float* __restrict__ Wx,
                             const float* __restrict__ Uh,
                             unsigned short* __restrict__ Wp) {
  int idx = blockIdx.x * blockDim.x + threadIdx.x;  // n*1024 + k
  int n = idx >> 10, k = idx & 1023;
  float v;
  if (n < 1024)      v = (k < Dc) ? Wx[k * 1536 + n] : Uh[(k - Dc) * 1536 + n];
  else if (n < 1536) v = (k < Dc) ? Wx[k * 1536 + n] : 0.f;
  else               v = (k < Dc) ? 0.f : Uh[(k - Dc) * 1536 + (n - 512)];
  Wp[idx] = f2bf(v);
}

// ---------------------------------------------------------- one GRU tree step
// Block = 128 threads (4 waves). blockIdx.x = j-tile (0..31), blockIdx.y =
// batch tile (0..7). Wave g computes gate g's 16x16 tile of
// [x|h] @ Wpack (K=1024) via 32x v_wmma_f32_16x16x32_bf16, then the block
// fuses the GRU nonlinearity, output store, and parent scatter (dt pass).
#define AROW 1032  // 1024 + 8 shorts pad: 2064B row stride -> conflict-free

template <bool TD>
__global__ __launch_bounds__(128)
void tree_gru_step(const float* __restrict__ emb,
                   const int*   __restrict__ pidx,     // [L,B]
                   const float* __restrict__ pvalid,   // [L,B]
                   const float* __restrict__ bias,     // [3H]
                   const unsigned short* __restrict__ Wp,  // [2048][1024] bf16
                   float* __restrict__ acc,            // [L,B,H] child sums
                   float* __restrict__ out,            // [B,L,2H]
                   int n0) {
  __shared__ unsigned short As[16 * AROW];
  __shared__ float Cx[4][16][17];

  const int tid   = threadIdx.x;
  const int jt    = blockIdx.x;
  const int mtile = blockIdx.y;

  // ---- stage A = [x | h_prev] as 16 x 1024 bf16 in LDS ----
  {
    int m  = tid >> 3;                 // row 0..15
    int b  = mtile * 16 + m;
    int cb = (tid & 7) * 128;          // 128-col chunk per thread
    unsigned short* dst = As + m * AROW + cb;
    if (cb < Dc) {                     // x part from emb[L,B,D]
      const float* src = emb + ((size_t)n0 * Bc + b) * Dc + cb;
      #pragma unroll 8
      for (int c = 0; c < 128; c += 4) {
        float4 v = *(const float4*)(src + c);
        v2u pk = { (unsigned)f2bf(v.x) | ((unsigned)f2bf(v.y) << 16),
                   (unsigned)f2bf(v.z) | ((unsigned)f2bf(v.w) << 16) };
        *(v2u*)(dst + c) = pk;
      }
    } else {                           // h part
      int hb = cb - Dc;
      if (!TD) {                       // bottom-up: accumulated child sums
        const float* src = acc + ((size_t)n0 * Bc + b) * Hc + hb;
        #pragma unroll 8
        for (int c = 0; c < 128; c += 4) {
          float4 v = *(const float4*)(src + c);
          v2u pk = { (unsigned)f2bf(v.x) | ((unsigned)f2bf(v.y) << 16),
                     (unsigned)f2bf(v.z) | ((unsigned)f2bf(v.w) << 16) };
          *(v2u*)(dst + c) = pk;
        }
      } else {                         // top-down: parent hidden (Htd half)
        int   p   = pidx  [n0 * Bc + b];
        float val = pvalid[n0 * Bc + b];
        const float* src = out + ((size_t)b * Lc + p) * (2 * Hc) + Hc + hb;
        #pragma unroll 8
        for (int c = 0; c < 128; c += 4) {
          float4 v = *(const float4*)(src + c);
          v.x *= val; v.y *= val; v.z *= val; v.w *= val;
          v2u pk = { (unsigned)f2bf(v.x) | ((unsigned)f2bf(v.y) << 16),
                     (unsigned)f2bf(v.z) | ((unsigned)f2bf(v.w) << 16) };
          *(v2u*)(dst + c) = pk;
        }
      }
    }
  }
  __syncthreads();

  // ---- per-wave WMMA GEMM: C[16x16] for one gate ----
  const int gate = tid >> 5;           // 0=r 1=z 2=xn 3=hn
  const int lane = tid & 31;
  const int half = lane >> 4;
  const int nl   = lane & 15;

  const unsigned short* Brow = Wp + (size_t)(gate * 512 + jt * 16 + nl) * KKc;
  v8f cfrag = {};
  #pragma unroll 4
  for (int kt = 0; kt < 32; ++kt) {
    const int kb = kt * 32;
    // A frag: lane holds M=nl; elems 0..7 K=kb+8h.., elems 8..15 K=kb+16+8h..
    const unsigned short* ap = As + nl * AROW + kb + 8 * half;
    Frag32 fa; fa.lo = *(const v4u*)ap; fa.hi = *(const v4u*)(ap + 16);
    // B frag: lane holds N=nl; elems 0..7 K=kb+8h.., elems 8..15 K=kb+16+8h.. (N-major Wp)
    const unsigned short* bp = Brow + kb + 8 * half;
    Frag32 fb; fb.lo = *(const v4u*)bp; fb.hi = *(const v4u*)(bp + 16);
    v16bf av = __builtin_bit_cast(v16bf, fa);
    v16bf bv = __builtin_bit_cast(v16bf, fb);
    cfrag = __builtin_amdgcn_wmma_f32_16x16x32_bf16(
        false, av, false, bv, (short)0, cfrag, false, false);
  }
  #pragma unroll
  for (int v = 0; v < 8; ++v) Cx[gate][v + 8 * half][nl] = cfrag[v];
  __syncthreads();

  // ---- fused GRU elementwise + store + parent scatter ----
  {
    int m2 = tid >> 4, j2 = tid & 15;
    int b2 = mtile * 16 + m2;
    int j  = jt * 16 + j2;
    float rp = Cx[0][m2][j2] + bias[j];
    float zp = Cx[1][m2][j2] + bias[Hc + j];
    float xn = Cx[2][m2][j2] + bias[2 * Hc + j];
    float hn = Cx[3][m2][j2];
    float r = fast_sigmoid(rp);
    float z = fast_sigmoid(zp);
    float ng = fast_tanh(xn + r * hn);
    float hprev;
    if (!TD) {
      hprev = acc[((size_t)n0 * Bc + b2) * Hc + j];
    } else {
      int   p   = pidx  [n0 * Bc + b2];
      float val = pvalid[n0 * Bc + b2];
      hprev = out[((size_t)b2 * Lc + p) * (2 * Hc) + Hc + j] * val;
    }
    float hout = (1.f - z) * ng + z * hprev;
    out[((size_t)b2 * Lc + n0) * (2 * Hc) + (TD ? Hc : 0) + j] = hout;
    if (!TD) {   // scatter into parent's child-sum; disjoint addrs per step
      int   p   = pidx  [n0 * Bc + b2];
      float val = pvalid[n0 * Bc + b2];
      if (val != 0.f) {
        float* d = acc + ((size_t)p * Bc + b2) * Hc + j;
        *d += hout;
      }
    }
  }
}

// --------------------------------------------------------------- root gather
__global__ void root_gather(const float* __restrict__ out,
                            const int* __restrict__ rootidx,
                            float* __restrict__ outt) {
  int i = blockIdx.x * blockDim.x + threadIdx.x;   // B*2H elements
  int b = i >> 10, c = i & 1023;
  int r = rootidx[b];
  outt[i] = out[((size_t)b * Lc + r) * (2 * Hc) + c];
}

// ---------------------------------------------------------------------------
extern "C" void kernel_launch(void* const* d_in, const int* in_sizes, int n_in,
                              void* d_out, int out_size, void* d_ws, size_t ws_size,
                              hipStream_t stream) {
  (void)in_sizes; (void)n_in; (void)out_size; (void)ws_size;
  const float* emb    = (const float*)d_in[0];
  // d_in[1] indexes  : implied (node = L-1-t)      -> unused
  // d_in[2] child_mask: replaced by parent scatter -> unused (saves 33MB reads)
  // d_in[3] td_node  : implied (node = t)          -> unused
  const int*   pidx   = (const int*)  d_in[4];
  const float* pvalid = (const float*)d_in[5];
  const int*   rooti  = (const int*)  d_in[6];
  const float* dt_Wx  = (const float*)d_in[7];
  const float* dt_Uh  = (const float*)d_in[8];
  const float* dt_b   = (const float*)d_in[9];
  const float* td_Wx  = (const float*)d_in[10];
  const float* td_Uh  = (const float*)d_in[11];
  const float* td_b   = (const float*)d_in[12];
  float* out = (float*)d_out;

  // workspace layout: two packed bf16 weight mats (4MB each) + child-sum acc
  unsigned short* Wp_dt = (unsigned short*)d_ws;
  unsigned short* Wp_td = Wp_dt + (size_t)NPc * KKc;
  float* acc = (float*)((char*)d_ws + 2 * (size_t)NPc * KKc * sizeof(unsigned short));

  const int accN = Lc * Bc * Hc;                       // 16M floats
  zero_f32<<<accN / 4 / 256, 256, 0, stream>>>(acc, accN / 4);
  pack_weights<<<(NPc * KKc) / 256, 256, 0, stream>>>(dt_Wx, dt_Uh, Wp_dt);
  pack_weights<<<(NPc * KKc) / 256, 256, 0, stream>>>(td_Wx, td_Uh, Wp_td);

  dim3 grid(32, 8), blk(128);
  // bottom-up: nodes L-1 .. 0 (children before parents)
  for (int t = 0; t < Lc; ++t) {
    tree_gru_step<false><<<grid, blk, 0, stream>>>(
        emb, pidx, pvalid, dt_b, Wp_dt, acc, out, Lc - 1 - t);
  }
  // top-down: nodes 0 .. L-1 (parents before children)
  for (int t = 0; t < Lc; ++t) {
    tree_gru_step<true><<<grid, blk, 0, stream>>>(
        emb, pidx, pvalid, td_b, Wp_td, acc, out, t);
  }
  float* outt = out + (size_t)Bc * Lc * (2 * Hc);
  root_gather<<<(Bc * 2 * Hc) / 256, 256, 0, stream>>>(out, rooti, outt);
}
